// ContrastiveLossLayer_10565619548958
// MI455X (gfx1250) — matrix operand order repaired and checked by reading, not compiled, not run on hardware
//
#include <hip/hip_runtime.h>
#include <math.h>

#define B_DIM 4096
#define D_DIM 1024
#define EPSF  1e-7f

typedef __attribute__((ext_vector_type(16))) __bf16 bf16x16;
typedef __attribute__((ext_vector_type(8)))  __bf16 bf16x8;
typedef __attribute__((ext_vector_type(4)))  __bf16 bf16x4;
typedef __attribute__((ext_vector_type(8)))  float  f32x8;

// ---------------- zero init (workspace is poisoned by harness) ----------------
__global__ void zero_f32_kernel(float* __restrict__ p, int n) {
    int i = blockIdx.x * blockDim.x + threadIdx.x;
    if (i < n) p[i] = 0.0f;
}

// ---------------- row L2-normalize fp32 -> bf16 ----------------
// grid: 2*B_DIM blocks of 256 threads; block b < B_DIM handles left row b,
// block b >= B_DIM handles right row b-B_DIM. Each thread owns 4 contiguous floats.
__global__ __launch_bounds__(256) void norm_kernel(const float* __restrict__ left,
                                                   const float* __restrict__ right,
                                                   __bf16* __restrict__ lnb,
                                                   __bf16* __restrict__ rnb) {
    const int  row     = blockIdx.x & (B_DIM - 1);
    const bool isRight = blockIdx.x >= B_DIM;
    const float* src = (isRight ? right : left) + (size_t)row * D_DIM;
    __bf16*      dst = (isRight ? rnb  : lnb ) + (size_t)row * D_DIM;

    const int t = threadIdx.x;            // 0..255, 256*4 = 1024 = D_DIM
    float4 v = ((const float4*)src)[t];
    float ss = v.x * v.x + v.y * v.y + v.z * v.z + v.w * v.w;
#pragma unroll
    for (int m = 1; m < 32; m <<= 1) ss += __shfl_xor(ss, m, 32);

    __shared__ float red[8];
    if ((t & 31) == 0) red[t >> 5] = ss;
    __syncthreads();
    float tot = 0.0f;
#pragma unroll
    for (int w = 0; w < 8; ++w) tot += red[w];

    const float invn = 1.0f / sqrtf(fmaxf(tot, EPSF));   // matches x/sqrt(max(ss,eps))
    bf16x4 o;
    o[0] = (__bf16)(v.x * invn);
    o[1] = (__bf16)(v.y * invn);
    o[2] = (__bf16)(v.z * invn);
    o[3] = (__bf16)(v.w * invn);
    ((bf16x4*)dst)[t] = o;
}

// ---------------- fused WMMA GEMM + exp + row/col/diag reductions ----------------
// Block = 256 threads = 8 waves arranged 2(M) x 4(N); wave tile = 64x64 (4x4 WMMA).
// Block tile = 128(M) x 256(N). Grid = (4096/128, 4096/256) = (32, 16).
__global__ __launch_bounds__(256) void gemm_softmax_kernel(
        const __bf16* __restrict__ lnb, const __bf16* __restrict__ rnb,
        const float* __restrict__ temperature,
        float* __restrict__ rowsum, float* __restrict__ colsum,
        float* __restrict__ diagE) {
    const int lane  = threadIdx.x & 31;
    const int wave  = threadIdx.x >> 5;       // 0..7
    const int waveM = wave & 1;               // 2 waves along M
    const int waveN = wave >> 1;              // 4 waves along N
    const int rowBase = blockIdx.x * 128 + waveM * 64;
    const int colBase = blockIdx.y * 256 + waveN * 64;
    const int lm = lane & 15;                 // lane within half
    const int lh = lane >> 4;                 // half-wave select

    f32x8 acc[4][4];
#pragma unroll
    for (int i = 0; i < 4; ++i)
#pragma unroll
        for (int j = 0; j < 4; ++j)
            acc[i][j] = (f32x8){0.f, 0.f, 0.f, 0.f, 0.f, 0.f, 0.f, 0.f};

    // Per-lane base pointers; A and B share one contiguous k order (k = 16h + s):
    //   A (16x32, MxK): lane L -> row M=L%16; K chunk  [ (L>>4)*16 .. +15 ]
    //   B (32x16, KxN): lane L -> col N=L%16; K chunk  [ (L>>4)*16 .. +15 ]  (B[k][n]=rn[n][k])
    const __bf16* aPtr[4];
    const __bf16* bPtr[4];
#pragma unroll
    for (int i = 0; i < 4; ++i)
        aPtr[i] = lnb + (size_t)(rowBase + i * 16 + lm) * D_DIM + lh * 16;
#pragma unroll
    for (int j = 0; j < 4; ++j)
        bPtr[j] = rnb + (size_t)(colBase + j * 16 + lm) * D_DIM + lh * 16;

    for (int kb = 0; kb < D_DIM; kb += 32) {
        bf16x16 a[4], b[4];
#pragma unroll
        for (int i = 0; i < 4; ++i) {
            a[i] = *(const bf16x16*)(aPtr[i] + kb);
        }
#pragma unroll
        for (int j = 0; j < 4; ++j)
            b[j] = *(const bf16x16*)(bPtr[j] + kb);
#pragma unroll
        for (int i = 0; i < 4; ++i)
#pragma unroll
            for (int j = 0; j < 4; ++j)
                acc[i][j] = __builtin_amdgcn_wmma_f32_16x16x32_bf16(
                    false, a[i], false, b[j], (short)0, acc[i][j], false, false);
    }

    // ---- epilogue: E = exp(scale * logit), then row/col/diag reductions ----
    const float scale = __expf(temperature[0]);
#pragma unroll
    for (int i = 0; i < 4; ++i)
#pragma unroll
        for (int j = 0; j < 4; ++j)
#pragma unroll
            for (int r = 0; r < 8; ++r)
                acc[i][j][r] = __expf(acc[i][j][r] * scale);

    // Column sums: element (M = r + lh*8, N = lm) in acc[..][j][r].
    // Sum over r and i in-lane, fold the two half-waves, one atomic per column.
#pragma unroll
    for (int j = 0; j < 4; ++j) {
        float cs = 0.0f;
#pragma unroll
        for (int i = 0; i < 4; ++i)
#pragma unroll
            for (int r = 0; r < 8; ++r) cs += acc[i][j][r];
        cs += __shfl_xor(cs, 16, 32);
        if (lane < 16) atomicAdd(&colsum[colBase + j * 16 + lane], cs);
    }

    // Row sums: sum over j in-lane, then reduce the 16 lanes of each half-wave.
#pragma unroll
    for (int i = 0; i < 4; ++i) {
#pragma unroll
        for (int r = 0; r < 8; ++r) {
            float t = acc[i][0][r] + acc[i][1][r] + acc[i][2][r] + acc[i][3][r];
            t += __shfl_xor(t, 1, 32);
            t += __shfl_xor(t, 2, 32);
            t += __shfl_xor(t, 4, 32);
            t += __shfl_xor(t, 8, 32);
            if (lm == 0) atomicAdd(&rowsum[rowBase + i * 16 + lh * 8 + r], t);
        }
    }

    // Diagonal exps: sub-tiles are 16-aligned in both M and N, so diagonal
    // elements only occur when the sub-tile bases coincide. Unique writer.
#pragma unroll
    for (int i = 0; i < 4; ++i)
#pragma unroll
        for (int j = 0; j < 4; ++j) {
            const int R = rowBase + i * 16, C = colBase + j * 16;
            if (R == C) {
#pragma unroll
                for (int r = 0; r < 8; ++r) {
                    const int m = lh * 8 + r;      // row within sub-tile
                    if (m == lm) diagE[R + m] = acc[i][j][r];  // m == n
                }
            }
        }
}

// ---------------- final scalar loss ----------------
__global__ __launch_bounds__(256) void loss_kernel(const float* __restrict__ rowsum,
                                                   const float* __restrict__ colsum,
                                                   const float* __restrict__ diagE,
                                                   float* __restrict__ out) {
    float sL = 0.0f, sR = 0.0f;
    for (int i = threadIdx.x; i < B_DIM; i += 256) {
        const float d = diagE[i];
        sL += d / rowsum[i];      // diag of prob_left (row softmax)
        sR += d / colsum[i];      // diag of prob_right (col softmax)
    }
#pragma unroll
    for (int m = 1; m < 32; m <<= 1) {
        sL += __shfl_xor(sL, m, 32);
        sR += __shfl_xor(sR, m, 32);
    }
    __shared__ float rl[8], rr[8];
    if ((threadIdx.x & 31) == 0) { rl[threadIdx.x >> 5] = sL; rr[threadIdx.x >> 5] = sR; }
    __syncthreads();
    if (threadIdx.x == 0) {
        float SL = 0.0f, SR = 0.0f;
#pragma unroll
        for (int w = 0; w < 8; ++w) { SL += rl[w]; SR += rr[w]; }
        const float C0 = logf(EPSF);          // log(eps)     (off-diagonal weight)
        const float C1 = logf(1.0f - EPSF);   // log(1 - eps) (diagonal weight)
        const float lossL = -(C1 * SL + C0 * ((float)B_DIM - SL));
        const float lossR = -(C1 * SR + C0 * ((float)B_DIM - SR));
        out[0] = (lossL + lossR) * (0.5f / (float)B_DIM);   // WEIGHT = 1.0
    }
}

extern "C" void kernel_launch(void* const* d_in, const int* in_sizes, int n_in,
                              void* d_out, int out_size, void* d_ws, size_t ws_size,
                              hipStream_t stream) {
    (void)in_sizes; (void)n_in; (void)out_size; (void)ws_size;
    const float* left  = (const float*)d_in[0];
    const float* right = (const float*)d_in[1];
    const float* temp  = (const float*)d_in[2];

    // workspace layout: lnb(8MB) | rnb(8MB) | rowsum(16KB) | colsum(16KB) | diagE(16KB)
    char* ws = (char*)d_ws;
    __bf16* lnb = (__bf16*)ws;
    __bf16* rnb = (__bf16*)(ws + (size_t)B_DIM * D_DIM * sizeof(__bf16));
    float* rowsum = (float*)(ws + (size_t)B_DIM * D_DIM * sizeof(__bf16) * 2);
    float* colsum = rowsum + B_DIM;
    float* diagE  = colsum + B_DIM;

    zero_f32_kernel<<<(3 * B_DIM + 255) / 256, 256, 0, stream>>>(rowsum, 3 * B_DIM);
    norm_kernel<<<2 * B_DIM, 256, 0, stream>>>(left, right, lnb, rnb);
    gemm_softmax_kernel<<<dim3(B_DIM / 128, B_DIM / 256), 256, 0, stream>>>(
        lnb, rnb, temp, rowsum, colsum, diagE);
    loss_kernel<<<1, 256, 0, stream>>>(rowsum, colsum, diagE, (float*)d_out);
}
